// EdgeUpdate_87505663689044
// MI455X (gfx1250) — compile-verified, matrix operand order checked by reading, not run on hardware
//
#include <hip/hip_runtime.h>

// ---------------------------------------------------------------------------
// EdgeUpdate (CGCNN-style 2-body + 3-body edge update), MI455X / gfx1250.
// bf16 WMMA GEMMs (c2@W2 : 10240x128x256, c3@W3 : 194560x640x256) with f32
// accumulation; GEMM3 A-tiles gathered with async Global->LDS DMA
// (global_load_async_to_lds_b128, ASYNCcnt); fused batch-norm + sigmoid*tanh.
// ---------------------------------------------------------------------------

typedef __attribute__((ext_vector_type(16))) __bf16 v16bf;
typedef __attribute__((ext_vector_type(8)))  __bf16 v8bf;
typedef __attribute__((ext_vector_type(4)))  __bf16 v4bf;
typedef __attribute__((ext_vector_type(8)))  float  v8f;

#define B_    4
#define AT    128
#define NBR   20
#define NK    19
#define NF    128
#define EF    128
#define OUTC  256
#define K3DIM 640
#define C2ROWS (B_*AT*NBR)        // 10240
#define C3ROWS (B_*AT*NBR*NK)     // 194560
#define EPS_  1e-5f

static __device__ __forceinline__ float sigmoid_(float x) {
    return 1.0f / (1.0f + __expf(-x));
}

// ---------------------------------------------------------------------------
__global__ void k_zero(float* __restrict__ p, int n) {
    int i = blockIdx.x * blockDim.x + threadIdx.x;
    if (i < n) p[i] = 0.0f;
}

// Pack a (K x 256) f32 weight matrix into per-lane contiguous bf16 fragments.
// Layout: dst[((s*16 + ct)*32 + lane)*16 + i], following ISA 7.12.2 for the
// 16-bit 32x16 B matrix: lanes 0-15 hold col=lane, K = {0..7, 16..23};
// lanes 16-31 hold col=lane-16, K = {8..15, 24..31} (per 32-wide k-step s).
__global__ void k_pack(const float* __restrict__ W, __bf16* __restrict__ dst, int ksteps) {
    int idx = blockIdx.x * blockDim.x + threadIdx.x;
    int total = ksteps * 16 * 32 * 16;
    if (idx >= total) return;
    int i  = idx & 15;
    int l  = (idx >> 4) & 31;
    int ct = (idx >> 9) & 15;
    int s  = idx >> 13;
    int col = ct * 16 + (l & 15);
    int grp = i >> 3, o = i & 7;
    int k = (l < 16) ? (s * 32 + (grp ? 16 + o : o))
                     : (s * 32 + (grp ? 24 + o : 8 + o));
    dst[idx] = (__bf16)W[(size_t)k * OUTC + col];
}

// c2[b,a,n,c] = node[b,a,c] * node[b, nbr_idx[b,a,n], c] * mask  -> bf16
__global__ void k_build_c2(const float* __restrict__ node,
                           const int* __restrict__ nbr_idx,
                           const int* __restrict__ nbr_mask,
                           __bf16* __restrict__ cs2) {
    int idx = blockIdx.x * blockDim.x + threadIdx.x;
    if (idx >= C2ROWS * NF) return;
    int c  = idx & 127;
    int rr = idx >> 7;           // (b*AT + a)*NBR + n
    int ba = rr / NBR;           // b*AT + a
    int b  = ba / AT;
    int j  = nbr_idx[rr];
    float m = (float)nbr_mask[rr];
    float v = node[(size_t)ba * NF + c] * node[(size_t)(b * AT + j) * NF + c] * m;
    cs2[idx] = (__bf16)v;
}

// GEMM2: h2 = cs2 (10240x128, bf16) @ W2pack -> f32, plus per-channel
// sum / sumsq accumulation for batch-norm. 640 blocks x 256 threads (8 waves,
// 2 column tiles per wave, 4 k-steps of 32).
__global__ void __launch_bounds__(256)
k_gemm2(const __bf16* __restrict__ cs2, const __bf16* __restrict__ w2p,
        float* __restrict__ h2, float* __restrict__ sum2, float* __restrict__ sumsq2) {
    __shared__ float Otile[16 * OUTC];
    const int tile = blockIdx.x;
    const int tid  = threadIdx.x;
    const int lane = tid & 31, wave = tid >> 5;
    const int ct0 = wave * 2, ct1 = wave * 2 + 1;
    v8f acc0 = {}; v8f acc1 = {};
    const __bf16* arow = cs2 + (size_t)(tile * 16 + (lane & 15)) * NF + (lane >> 4) * 8;
    #pragma unroll
    for (int s = 0; s < 4; ++s) {
        v8bf alo = *(const v8bf*)(arow + s * 32);
        v8bf ahi = *(const v8bf*)(arow + s * 32 + 16);
        v16bf a = __builtin_shufflevector(alo, ahi, 0,1,2,3,4,5,6,7,8,9,10,11,12,13,14,15);
        v16bf b0 = *(const v16bf*)(w2p + ((size_t)((s * 16 + ct0) * 32 + lane)) * 16);
        v16bf b1 = *(const v16bf*)(w2p + ((size_t)((s * 16 + ct1) * 32 + lane)) * 16);
        acc0 = __builtin_amdgcn_wmma_f32_16x16x32_bf16(false, a, false, b0, (short)0, acc0, false, false);
        acc1 = __builtin_amdgcn_wmma_f32_16x16x32_bf16(false, a, false, b1, (short)0, acc1, false, false);
    }
    const int rbase = (lane < 16) ? 0 : 8;
    const int c0 = ct0 * 16 + (lane & 15), c1 = ct1 * 16 + (lane & 15);
    #pragma unroll
    for (int v = 0; v < 8; ++v) {
        Otile[(rbase + v) * OUTC + c0] = acc0[v];
        Otile[(rbase + v) * OUTC + c1] = acc1[v];
    }
    __syncthreads();
    // tid == output channel: store rows + accumulate BN stats
    float csum = 0.0f, csq = 0.0f;
    #pragma unroll 4
    for (int i = 0; i < 16; ++i) {
        float x = Otile[i * OUTC + tid];
        csum += x; csq += x * x;
        h2[(size_t)(tile * 16 + i) * OUTC + tid] = x;
    }
    atomicAdd(&sum2[tid], csum);
    atomicAdd(&sumsq2[tid], csq);
}

// Fold BN stats into per-channel scale/shift: y = x*scale + shift
__global__ void k_fstats(const float* __restrict__ sum, const float* __restrict__ sumsq,
                         const float* __restrict__ g, const float* __restrict__ be,
                         float invN, int C, float* __restrict__ scale, float* __restrict__ shift) {
    int c = blockIdx.x * blockDim.x + threadIdx.x;
    if (c >= C) return;
    float m   = sum[c] * invN;
    float var = sumsq[c] * invN - m * m;
    float sc  = g[c] * rsqrtf(var + EPS_);
    scale[c] = sc;
    shift[c] = be[c] - m * sc;
}

// two_body = sigmoid(bn(gate)) * tanh(bn(ext))
__global__ void k_twobody(const float* __restrict__ h2, const float* __restrict__ scale2,
                          const float* __restrict__ shift2, float* __restrict__ twob) {
    int idx = blockIdx.x * blockDim.x + threadIdx.x;
    if (idx >= C2ROWS * NF) return;
    int c = idx & 127, r = idx >> 7;
    float gate = h2[(size_t)r * OUTC + c]       * scale2[c]       + shift2[c];
    float ext  = h2[(size_t)r * OUTC + 128 + c] * scale2[128 + c] + shift2[128 + c];
    twob[idx] = sigmoid_(gate) * tanhf(ext);
}

// GEMM3: c3 (194560 x 640) @ W3pack -> 256 cols.
// A-tile (16 rows x 640) = 80 contiguous 512B segments, pulled into LDS with
// async Global->LDS DMA (10 wave-instructions per wave), then converted f32
// -> bf16 once. mode 0: BN stats only. mode 1: normalize + sigmoid*tanh +
// atomic k-sum into tb3. 12160 blocks x 256 threads; 40 WMMAs per wave.
__global__ void __launch_bounds__(256)
k_gemm3(const float* __restrict__ node, const float* __restrict__ edge,
        const int* __restrict__ nbr_idx, const __bf16* __restrict__ w3p,
        const float* __restrict__ scale3, const float* __restrict__ shift3,
        float* __restrict__ sum3, float* __restrict__ sumsq3,
        float* __restrict__ tb3, int mode) {
    __shared__ float  Stage[16 * K3DIM];   // 40 KB f32 staging; reused as Otile
    __shared__ __bf16 Atile[16 * K3DIM];   // 20 KB bf16 A tile
    const int tile = blockIdx.x;
    const int tid  = threadIdx.x;
    const int lane = tid & 31, wave = tid >> 5;

    // ---- async-DMA the 80 row-segments [ni | nj | nk | eij | ekj] into LDS
    {
        const unsigned int stage_base = (unsigned int)(uintptr_t)(&Stage[0]);
        for (int q = wave * 10; q < wave * 10 + 10; ++q) {   // uniform per wave
            int i   = q / 5;        // tile row
            int seg = q % 5;
            int r  = tile * 16 + i;
            int k  = r % NK;
            int t1 = r / NK;
            int n  = t1 % NBR;
            int ba = t1 / NBR;      // b*AT + a
            int b  = ba / AT;
            int m  = (k < n) ? k : k + 1;
            int j  = nbr_idx[ba * NBR + n];
            const float* p;
            if      (seg == 0) p = node + (size_t)ba * NF;
            else if (seg == 1) p = node + (size_t)(b * AT + j) * NF;
            else if (seg == 2) {
                int kk = nbr_idx[(b * AT + j) * NBR + m];
                p = node + (size_t)(b * AT + kk) * NF;
            }
            else if (seg == 3) p = edge + (size_t)(ba * NBR + n) * EF;
            else               p = edge + (size_t)((b * AT + j) * NBR + m) * EF;
            unsigned long long ga = (unsigned long long)(uintptr_t)p
                                  + (unsigned long long)(lane * 16);
            unsigned int la = stage_base
                            + (unsigned int)((i * K3DIM + seg * 128) * 4 + lane * 16);
            // 32 lanes x 16B = one full 512B segment, straight into LDS
            asm volatile("global_load_async_to_lds_b128 %0, %1, off"
                         :: "v"(la), "v"(ga) : "memory");
        }
        asm volatile("s_wait_asynccnt 0x0" ::: "memory");
    }
    __syncthreads();

    // ---- convert staged f32 tile -> bf16 A tile (vectorized)
    for (int e = tid * 4; e < 16 * K3DIM; e += 256 * 4) {
        float4 vf = *(const float4*)&Stage[e];
        v4bf hv = { (__bf16)vf.x, (__bf16)vf.y, (__bf16)vf.z, (__bf16)vf.w };
        *(v4bf*)&Atile[e] = hv;
    }
    __syncthreads();

    // ---- WMMA main loop
    const int ct0 = wave * 2, ct1 = wave * 2 + 1;
    v8f acc0 = {}; v8f acc1 = {};
    const __bf16* arow = &Atile[(lane & 15) * K3DIM + (lane >> 4) * 8];
    for (int s = 0; s < 20; ++s) {
        v8bf alo = *(const v8bf*)(arow + s * 32);
        v8bf ahi = *(const v8bf*)(arow + s * 32 + 16);
        v16bf a = __builtin_shufflevector(alo, ahi, 0,1,2,3,4,5,6,7,8,9,10,11,12,13,14,15);
        const __bf16* bp0 = w3p + ((size_t)((s * 16 + ct0) * 32 + lane)) * 16;
        const __bf16* bp1 = w3p + ((size_t)((s * 16 + ct1) * 32 + lane)) * 16;
        if (s + 1 < 20) __builtin_prefetch(bp0 + 8192, 0, 3);  // next k-step weights
        v16bf b0 = *(const v16bf*)bp0;
        v16bf b1 = *(const v16bf*)bp1;
        acc0 = __builtin_amdgcn_wmma_f32_16x16x32_bf16(false, a, false, b0, (short)0, acc0, false, false);
        acc1 = __builtin_amdgcn_wmma_f32_16x16x32_bf16(false, a, false, b1, (short)0, acc1, false, false);
    }

    // ---- stage C tiles in LDS (reuse Stage as Otile; dead after conversion)
    float* Otile = Stage;
    const int rbase = (lane < 16) ? 0 : 8;
    const int c0 = ct0 * 16 + (lane & 15), c1 = ct1 * 16 + (lane & 15);
    #pragma unroll
    for (int v = 0; v < 8; ++v) {
        Otile[(rbase + v) * OUTC + c0] = acc0[v];
        Otile[(rbase + v) * OUTC + c1] = acc1[v];
    }
    __syncthreads();

    if (mode == 0) {
        // batch-norm statistics pass
        float csum = 0.0f, csq = 0.0f;
        #pragma unroll 4
        for (int i = 0; i < 16; ++i) {
            float x = Otile[i * OUTC + tid];
            csum += x; csq += x * x;
        }
        atomicAdd(&sum3[tid], csum);
        atomicAdd(&sumsq3[tid], csq);
    } else {
        // normalize + sigmoid*tanh + k-reduction into tb3
        int c = tid & 127, half = tid >> 7;
        float sc_g = scale3[c],       sh_g = shift3[c];
        float sc_e = scale3[128 + c], sh_e = shift3[128 + c];
        #pragma unroll
        for (int i = half * 8; i < half * 8 + 8; ++i) {
            float gate = Otile[i * OUTC + c]       * sc_g + sh_g;
            float ext  = Otile[i * OUTC + 128 + c] * sc_e + sh_e;
            float act  = sigmoid_(gate) * tanhf(ext);
            int r = tile * 16 + i;
            int g = r / NK;          // (b*AT + a)*NBR + n
            atomicAdd(&tb3[(size_t)g * EF + c], act);
        }
    }
}

// Per-channel BN stats over tb3 (10240 rows x 128 channels) -> scale/shift
__global__ void __launch_bounds__(256)
k_tb3stats(const float* __restrict__ tb3, const float* __restrict__ gs,
           const float* __restrict__ bes, float* __restrict__ scaleS,
           float* __restrict__ shiftS) {
    __shared__ float s_sum[256], s_sq[256];
    int c   = blockIdx.x;      // channel 0..127
    int tid = threadIdx.x;
    float lsum = 0.0f, lsq = 0.0f;
    for (int g = tid; g < C2ROWS; g += 256) {
        float x = tb3[(size_t)g * EF + c];
        lsum += x; lsq += x * x;
    }
    s_sum[tid] = lsum; s_sq[tid] = lsq;
    __syncthreads();
    for (int ofs = 128; ofs > 0; ofs >>= 1) {
        if (tid < ofs) { s_sum[tid] += s_sum[tid + ofs]; s_sq[tid] += s_sq[tid + ofs]; }
        __syncthreads();
    }
    if (tid == 0) {
        float invN = 1.0f / (float)C2ROWS;
        float m   = s_sum[0] * invN;
        float var = s_sq[0] * invN - m * m;
        float sc  = gs[c] * rsqrtf(var + EPS_);
        scaleS[c] = sc;
        shiftS[c] = bes[c] - m * sc;
    }
}

// out = tanh(edge + two_body + bn(three_body))
__global__ void k_final(const float* __restrict__ edge, const float* __restrict__ twob,
                        const float* __restrict__ tb3, const float* __restrict__ scaleS,
                        const float* __restrict__ shiftS, float* __restrict__ out) {
    int idx = blockIdx.x * blockDim.x + threadIdx.x;
    if (idx >= C2ROWS * EF) return;
    int c = idx & 127;
    out[idx] = tanhf(edge[idx] + twob[idx] + tb3[idx] * scaleS[c] + shiftS[c]);
}

// ---------------------------------------------------------------------------
extern "C" void kernel_launch(void* const* d_in, const int* in_sizes, int n_in,
                              void* d_out, int out_size, void* d_ws, size_t ws_size,
                              hipStream_t stream) {
    const float* node     = (const float*)d_in[0];
    const float* edge     = (const float*)d_in[1];
    const int*   nbr_idx  = (const int*)d_in[2];
    const int*   nbr_mask = (const int*)d_in[3];
    const float* W2  = (const float*)d_in[4];
    const float* W3  = (const float*)d_in[6];
    const float* g2  = (const float*)d_in[8];
    const float* be2 = (const float*)d_in[9];
    const float* g3  = (const float*)d_in[10];
    const float* be3 = (const float*)d_in[11];
    const float* gs  = (const float*)d_in[12];
    const float* bes = (const float*)d_in[13];
    float* out = (float*)d_out;

    // workspace carve-up (256B aligned)
    char* ws = (char*)d_ws;
    size_t off = 0;
    auto carve = [&](size_t bytes) -> void* {
        void* p = ws + off;
        off += (bytes + 255) & ~(size_t)255;
        return p;
    };
    __bf16* w2p  = (__bf16*)carve((size_t)4  * 16 * 32 * 16 * sizeof(__bf16)); //  64 KB
    __bf16* w3p  = (__bf16*)carve((size_t)20 * 16 * 32 * 16 * sizeof(__bf16)); // 320 KB
    __bf16* cs2  = (__bf16*)carve((size_t)C2ROWS * NF * sizeof(__bf16));       // 2.5 MB
    float*  h2   = (float*) carve((size_t)C2ROWS * OUTC * sizeof(float));      //  10 MB
    float*  twob = (float*) carve((size_t)C2ROWS * NF * sizeof(float));        //   5 MB
    float*  tb3  = (float*) carve((size_t)C2ROWS * EF * sizeof(float));        //   5 MB
    float*  stats = (float*)carve((size_t)2304 * sizeof(float));
    float* sum2   = stats;        float* sumsq2 = stats + 256;
    float* sum3   = stats + 512;  float* sumsq3 = stats + 768;
    float* scale2 = stats + 1024; float* shift2 = stats + 1280;
    float* scale3 = stats + 1536; float* shift3 = stats + 1792;
    float* scaleS = stats + 2048; float* shiftS = stats + 2176;

    const int T = 256;
    // 0) zero accumulators (atomics accumulate into these)
    k_zero<<<(C2ROWS * EF + T - 1) / T, T, 0, stream>>>(tb3, C2ROWS * EF);
    k_zero<<<(2304 + T - 1) / T, T, 0, stream>>>(stats, 2304);
    // 1) pack weights to bf16 WMMA-B fragments
    k_pack<<<(4  * 16 * 32 * 16 + T - 1) / T, T, 0, stream>>>(W2, w2p, 4);
    k_pack<<<(20 * 16 * 32 * 16 + T - 1) / T, T, 0, stream>>>(W3, w3p, 20);
    // 2) two-body path
    k_build_c2<<<(C2ROWS * NF + T - 1) / T, T, 0, stream>>>(node, nbr_idx, nbr_mask, cs2);
    k_gemm2<<<C2ROWS / 16, T, 0, stream>>>(cs2, w2p, h2, sum2, sumsq2);
    k_fstats<<<1, 256, 0, stream>>>(sum2, sumsq2, g2, be2, 1.0f / (float)C2ROWS, 256, scale2, shift2);
    k_twobody<<<(C2ROWS * NF + T - 1) / T, T, 0, stream>>>(h2, scale2, shift2, twob);
    // 3) three-body path: stats pass, finalize, fused recompute pass
    k_gemm3<<<C3ROWS / 16, T, 0, stream>>>(node, edge, nbr_idx, w3p,
                                           scale3, shift3, sum3, sumsq3, tb3, 0);
    k_fstats<<<1, 256, 0, stream>>>(sum3, sumsq3, g3, be3, 1.0f / (float)C3ROWS, 256, scale3, shift3);
    k_gemm3<<<C3ROWS / 16, T, 0, stream>>>(node, edge, nbr_idx, w3p,
                                           scale3, shift3, sum3, sumsq3, tb3, 1);
    // 4) BN on three_body, final combine
    k_tb3stats<<<128, 256, 0, stream>>>(tb3, gs, bes, scaleS, shiftS);
    k_final<<<(C2ROWS * EF + T - 1) / T, T, 0, stream>>>(edge, twob, tb3, scaleS, shiftS, out);
}
